// MoE_63307817943543
// MI455X (gfx1250) — matrix operand order repaired and checked by reading, not compiled, not run on hardware
//
#include <hip/hip_runtime.h>
#include <math.h>

#define NTOK 2048   // B*L
#define D_ 1024
#define HID_ 2048
#define E_ 8
#define MT32 64     // max 32-token tiles per expert (2048/32)

typedef float v8f    __attribute__((ext_vector_type(8)));
typedef __bf16 v16bf __attribute__((ext_vector_type(16)));
typedef unsigned uint4v __attribute__((ext_vector_type(4)));
typedef unsigned short u16;

union V16 { uint4v q[2]; v16bf v; };

__device__ __forceinline__ u16 f2bf(float f) {
    unsigned u = __float_as_uint(f);
    unsigned r = u + 0x7FFFu + ((u >> 16) & 1u);   // round-to-nearest-even
    return (u16)(r >> 16);
}

__device__ __forceinline__ float gelu_tanh(float x) {
    float x3 = x * x * x;
    return 0.5f * x * (1.0f + tanhf(0.7978845608028654f * (x + 0.044715f * x3)));
}

// 16B-chunk XOR swizzle: spreads 16 rows across all 64 LDS banks
__device__ __forceinline__ int swz(int j, int m) {
    return (j & ~15) | ((j ^ m) & 15);
}

__device__ __forceinline__ v8f vzero() {
    return (v8f){0.f, 0.f, 0.f, 0.f, 0.f, 0.f, 0.f, 0.f};
}

// async 16B global->LDS copy (per-lane addresses); tracked by ASYNCcnt
__device__ __forceinline__ void async_ld16(unsigned lds_off, const void* gaddr) {
    asm volatile("global_load_async_to_lds_b128 %0, %1, off"
                 :: "v"(lds_off), "v"(gaddr) : "memory");
}

template <int N>
__device__ __forceinline__ void wait_async() {
    asm volatile("s_wait_asynccnt %0" :: "i"(N) : "memory");
}

__device__ __forceinline__ unsigned lds_off32(const void* p) {
    return (unsigned)(size_t)p;   // LDS flat address: low 32 bits are the LDS byte offset
}

// ---------------------------------------------------------------- zero out + counters
__global__ __launch_bounds__(256) void k_zero(uint4v* __restrict__ out, size_t n4,
                                              int* __restrict__ counts) {
    size_t i = (size_t)blockIdx.x * blockDim.x + threadIdx.x;
    uint4v z = {0u, 0u, 0u, 0u};
    for (; i < n4; i += (size_t)gridDim.x * blockDim.x) out[i] = z;
    if (blockIdx.x == 0 && threadIdx.x < E_) counts[threadIdx.x] = 0;
}

// ---------------------------------------------------------------- fp32 -> bf16 (layout preserved)
__global__ __launch_bounds__(256) void k_cvt(const float* __restrict__ src,
                                             u16* __restrict__ dst, size_t n) {
    size_t i = ((size_t)blockIdx.x * blockDim.x + threadIdx.x) * 4;
    size_t stride = (size_t)gridDim.x * blockDim.x * 4;
    for (; i < n; i += stride) {
        float4 f = *(const float4*)(src + i);
        uint2 o;
        o.x = (unsigned)f2bf(f.x) | ((unsigned)f2bf(f.y) << 16);
        o.y = (unsigned)f2bf(f.z) | ((unsigned)f2bf(f.w) << 16);
        *(uint2*)(dst + i) = o;
    }
}

// ---------------------------------------------------------------- linear_w [E,H,D] -> bf16 [E,D,H]
__global__ __launch_bounds__(256) void k_transpose_w2(const float* __restrict__ lw,
                                                      u16* __restrict__ w2t) {
    __shared__ float tile[32][33];
    int e  = blockIdx.x / (HID_ / 32);
    int h0 = (blockIdx.x % (HID_ / 32)) * 32;
    int d0 = blockIdx.y * 32;
    int c  = threadIdx.x & 31;
    int r0 = threadIdx.x >> 5;
    const float* src = lw + ((size_t)e * HID_ + h0) * D_ + d0;
#pragma unroll
    for (int p = 0; p < 4; ++p) {
        int r = r0 + p * 8;
        tile[r][c] = src[(size_t)r * D_ + c];
    }
    __syncthreads();
    u16* dst = w2t + ((size_t)e * D_ + d0) * HID_ + h0;
#pragma unroll
    for (int p = 0; p < 4; ++p) {
        int r = r0 + p * 8;
        dst[(size_t)r * HID_ + c] = f2bf(tile[c][r]);
    }
}

// ---------------------------------------------------------------- router: 1 wave per token
__global__ __launch_bounds__(256) void k_router(const float* __restrict__ x,
                                                const float* __restrict__ rscale,
                                                const float* __restrict__ rw,   // [D,E]
                                                const float* __restrict__ pes,  // [E]
                                                float* __restrict__ combine,    // [N,E]
                                                int* __restrict__ counts,       // [E]
                                                int* __restrict__ idxlist,      // [E,N]
                                                u16* __restrict__ xb) {         // [N,D] bf16
    int wave = threadIdx.x >> 5, lane = threadIdx.x & 31;
    int t = blockIdx.x * 8 + wave;
    const float* xr = x + (size_t)t * D_;
    float xv[32];
    float ss = 0.f;
#pragma unroll
    for (int i = 0; i < 32; ++i) {
        float v = xr[i * 32 + lane];
        xv[i] = v;
        ss += v * v;
        xb[(size_t)t * D_ + i * 32 + lane] = f2bf(v);
    }
#pragma unroll
    for (int m = 16; m > 0; m >>= 1) ss += __shfl_xor(ss, m, 32);
    float scale = __frsqrt_rn(ss * (1.0f / D_) + 1e-6f) * __frsqrt_rn((float)D_);
    float acc[E_];
#pragma unroll
    for (int e = 0; e < E_; ++e) acc[e] = 0.f;
#pragma unroll
    for (int i = 0; i < 32; ++i) {
        int d = i * 32 + lane;
        float r = xv[i] * scale * rscale[d];
        const float4* wr = (const float4*)(rw + (size_t)d * E_);
        float4 w0 = wr[0], w1 = wr[1];
        acc[0] += r * w0.x; acc[1] += r * w0.y; acc[2] += r * w0.z; acc[3] += r * w0.w;
        acc[4] += r * w1.x; acc[5] += r * w1.y; acc[6] += r * w1.z; acc[7] += r * w1.w;
    }
#pragma unroll
    for (int e = 0; e < E_; ++e) {
#pragma unroll
        for (int m = 16; m > 0; m >>= 1) acc[e] += __shfl_xor(acc[e], m, 32);
    }
    if (lane == 0) {
        float mx = acc[0];
#pragma unroll
        for (int e = 1; e < E_; ++e) mx = fmaxf(mx, acc[e]);
        float p[E_];
#pragma unroll
        for (int e = 0; e < E_; ++e) p[e] = __expf(acc[e] - mx);
        int i0 = 0;
#pragma unroll
        for (int e = 1; e < E_; ++e) if (acc[e] > acc[i0]) i0 = e;
        int i1 = (i0 == 0) ? 1 : 0;
#pragma unroll
        for (int e = 0; e < E_; ++e) if (e != i0 && acc[e] > acc[i1]) i1 = e;
        float renorm = p[i0] + p[i1];
        float c0 = p[i0] / renorm * pes[i0];
        float c1 = p[i1] / renorm * pes[i1];
        float row[E_];
#pragma unroll
        for (int e = 0; e < E_; ++e) row[e] = 0.f;
        row[i0] = c0; row[i1] = c1;
#pragma unroll
        for (int e = 0; e < E_; ++e) combine[(size_t)t * E_ + e] = row[e];
        int s0 = atomicAdd(&counts[i0], 1); idxlist[i0 * NTOK + s0] = t;
        int s1 = atomicAdd(&counts[i1], 1); idxlist[i1 * NTOK + s1] = t;
    }
}

// ---------------------------------------------------------------- pass 1: act = gelu(xW0^T)*(xW1^T)
// block: (expert, 32-token tile, 512-h chunk). M=32 -> B fragments reused by 2 A fragments.
__global__ __launch_bounds__(256, 1) void k_act(const u16* __restrict__ xb,   // [N,D]
                                                const u16* __restrict__ wgb,  // [E,2,H,D]
                                                const int* __restrict__ counts,
                                                const int* __restrict__ idxlist,
                                                u16* __restrict__ act_ws) {   // [E*N, H]
    __shared__ u16 sX[32 * 1024];   // 64KB, XOR-swizzled 16B chunks (128 chunks/row)

    int e    = blockIdx.x >> 6;
    int tile = blockIdx.x & (MT32 - 1);
    int cnt  = counts[e];
    if (tile * 32 >= cnt) return;          // uniform exit
    int hc   = blockIdx.y;                  // h base = hc*512

    int tid = threadIdx.x, wave = tid >> 5, lane = tid & 31;
    int lgrp = lane >> 4, lcol = lane & 15;

    // stage x tile via async global->LDS (8 threads/row, 16 chunks each, swizzled)
    {
        int row = tid >> 3, c0 = tid & 7;
        int slot = tile * 32 + row;
        int tok = idxlist[e * NTOK + min(slot, cnt - 1)];
        const u16* src = xb + (size_t)tok * D_;
        unsigned dlds = lds_off32(sX + row * 1024);
#pragma unroll
        for (int i = 0; i < 16; ++i) {
            int j = c0 + 8 * i;
            async_ld16(dlds + (unsigned)swz(j, row) * 16u, src + j * 8);
        }
    }
    wait_async<0>();
    __syncthreads();

    const u16* w0e = wgb + ((size_t)e * 2 + 0) * HID_ * D_;
    const u16* w1e = wgb + ((size_t)e * 2 + 1) * HID_ * D_;
    const uint4v* sXq = (const uint4v*)sX;
    int m0 = lcol, m1 = 16 + lcol;          // A rows for the two M-tiles

#pragma unroll
    for (int nt = 0; nt < 4; ++nt) {
        int habs = hc * 512 + wave * 64 + nt * 16;
        v8f c00 = vzero(), c01 = vzero(), c10 = vzero(), c11 = vzero();
        const u16* b0p = w0e + (size_t)(habs + lcol) * D_ + 8 * lgrp;
        const u16* b1p = w1e + (size_t)(habs + lcol) * D_ + 8 * lgrp;
#pragma unroll 4
        for (int k0 = 0; k0 < D_; k0 += 32) {
            __builtin_prefetch(b0p + k0 + 256, 0, 3);
            __builtin_prefetch(b1p + k0 + 256, 0, 3);
            int jb = k0 >> 3;
            int j1 = jb + lgrp, j2 = jb + 2 + lgrp;
            V16 a0, a1, b0, b1;
            a0.q[0] = sXq[m0 * 128 + swz(j1, m0)];
            a0.q[1] = sXq[m0 * 128 + swz(j2, m0)];
            a1.q[0] = sXq[m1 * 128 + swz(j1, m1)];
            a1.q[1] = sXq[m1 * 128 + swz(j2, m1)];
            b0.q[0] = *(const uint4v*)(b0p + k0);
            b0.q[1] = *(const uint4v*)(b0p + k0 + 16);
            b1.q[0] = *(const uint4v*)(b1p + k0);
            b1.q[1] = *(const uint4v*)(b1p + k0 + 16);
            c00 = __builtin_amdgcn_wmma_f32_16x16x32_bf16(false, a0.v, false, b0.v, (short)0, c00, false, false);
            c10 = __builtin_amdgcn_wmma_f32_16x16x32_bf16(false, a1.v, false, b0.v, (short)0, c10, false, false);
            c01 = __builtin_amdgcn_wmma_f32_16x16x32_bf16(false, a0.v, false, b1.v, (short)0, c01, false, false);
            c11 = __builtin_amdgcn_wmma_f32_16x16x32_bf16(false, a1.v, false, b1.v, (short)0, c11, false, false);
        }
        // epilogue: act rows (coalesced b16 stores along h)
        size_t rowbase = (size_t)(e * NTOK + tile * 32);
#pragma unroll
        for (int r = 0; r < 8; ++r) {
            int mA = r + 8 * lgrp;          // C layout: M = r + 8*(lane/16), N = lane%16
            float g0 = gelu_tanh(c00[r]) * c01[r];
            act_ws[(rowbase + mA) * HID_ + habs + lcol] = f2bf(g0);
            float g1 = gelu_tanh(c10[r]) * c11[r];
            act_ws[(rowbase + 16 + mA) * HID_ + habs + lcol] = f2bf(g1);
        }
    }
}

// ---------------------------------------------------------------- pass 2: out += combine * (act @ W2)
// block: (expert, 32-token tile, 512-d half).
// Double-buffered async LDS staging of act chunks overlapped with WMMA.
__global__ __launch_bounds__(256, 2) void k_down(const u16* __restrict__ act_ws, // [E*N, H]
                                                 const u16* __restrict__ w2t,    // [E,D,H]
                                                 const float* __restrict__ combine,
                                                 const int* __restrict__ counts,
                                                 const int* __restrict__ idxlist,
                                                 float* __restrict__ out) {
    __shared__ u16 sA[2][32 * 512];    // 2 x 32KB act chunks, swizzled (64 chunks/row)

    int e    = blockIdx.x >> 6;
    int tile = blockIdx.x & (MT32 - 1);
    int cnt  = counts[e];
    if (tile * 32 >= cnt) return;
    int dbase = blockIdx.y * 512;

    int tid = threadIdx.x, wave = tid >> 5, lane = tid & 31;
    int lgrp = lane >> 4, lcol = lane & 15;

    const u16* w2e  = w2t + (size_t)e * D_ * HID_;
    const u16* arow = act_ws + (size_t)(e * NTOK + tile * 32) * HID_;
    int m0 = lcol, m1 = 16 + lcol;

    int srow = tid >> 3, sc0 = tid & 7;     // staging: 8 threads/row, 8 chunks each
    const u16* srcrow = arow + (size_t)srow * HID_;
    unsigned slds0 = lds_off32(&sA[0][srow * 512]);
    unsigned slds1 = lds_off32(&sA[1][srow * 512]);

    // issue chunk 0 DMA
#pragma unroll
    for (int i = 0; i < 8; ++i) {
        int j = sc0 + 8 * i;
        async_ld16(slds0 + (unsigned)swz(j, srow) * 16u, srcrow + j * 8);
    }

    v8f acc[4][2];
#pragma unroll
    for (int nt = 0; nt < 4; ++nt) { acc[nt][0] = vzero(); acc[nt][1] = vzero(); }

#pragma unroll
    for (int kc = 0; kc < 4; ++kc) {
        if (kc < 3) {   // issue next chunk DMA into the other buffer
            unsigned dl = ((kc + 1) & 1) ? slds1 : slds0;
            const u16* s = srcrow + (kc + 1) * 512;
#pragma unroll
            for (int i = 0; i < 8; ++i) {
                int j = sc0 + 8 * i;
                async_ld16(dl + (unsigned)swz(j, srow) * 16u, s + j * 8);
            }
            wait_async<8>();   // in-order completion: chunk kc landed, chunk kc+1 in flight
        } else {
            wait_async<0>();
        }
        __syncthreads();       // chunk kc visible to all waves

        const uint4v* sAq = (const uint4v*)sA[kc & 1];
#pragma unroll
        for (int nt = 0; nt < 4; ++nt) {
            int dabs = dbase + wave * 64 + nt * 16;
            const u16* bp = w2e + (size_t)(dabs + lcol) * HID_ + kc * 512 + 8 * lgrp;
            v8f cA = acc[nt][0], cB = acc[nt][1];
#pragma unroll 4
            for (int k0 = 0; k0 < 512; k0 += 32) {
                __builtin_prefetch(bp + k0 + 256, 0, 3);
                int jb = k0 >> 3;
                int j1 = jb + lgrp, j2 = jb + 2 + lgrp;
                V16 a0, a1, b;
                a0.q[0] = sAq[m0 * 64 + swz(j1, m0)];
                a0.q[1] = sAq[m0 * 64 + swz(j2, m0)];
                a1.q[0] = sAq[m1 * 64 + swz(j1, m1)];
                a1.q[1] = sAq[m1 * 64 + swz(j2, m1)];
                b.q[0] = *(const uint4v*)(bp + k0);
                b.q[1] = *(const uint4v*)(bp + k0 + 16);
                cA = __builtin_amdgcn_wmma_f32_16x16x32_bf16(false, a0.v, false, b.v, (short)0, cA, false, false);
                cB = __builtin_amdgcn_wmma_f32_16x16x32_bf16(false, a1.v, false, b.v, (short)0, cB, false, false);
            }
            acc[nt][0] = cA; acc[nt][1] = cB;
        }
        __syncthreads();       // chunk kc consumed; its buffer may be overwritten next round
    }

    // epilogue: combine weights / tokens straight from L2 (once per block)
    int   tA[8], tB[8];
    float cwA[8], cwB[8];
#pragma unroll
    for (int r = 0; r < 8; ++r) {
        int mA = r + 8 * lgrp;
        int slotA = tile * 32 + mA;
        int slotB = slotA + 16;
        tA[r] = idxlist[e * NTOK + min(slotA, cnt - 1)];
        tB[r] = idxlist[e * NTOK + min(slotB, cnt - 1)];
        cwA[r] = (slotA < cnt) ? combine[(size_t)tA[r] * E_ + e] : 0.f;
        cwB[r] = (slotB < cnt) ? combine[(size_t)tB[r] * E_ + e] : 0.f;
    }
#pragma unroll
    for (int nt = 0; nt < 4; ++nt) {
        int dabs = dbase + wave * 64 + nt * 16;
#pragma unroll
        for (int r = 0; r < 8; ++r) {
            atomicAdd(out + (size_t)tA[r] * D_ + dabs + lcol, acc[nt][0][r] * cwA[r]);
            atomicAdd(out + (size_t)tB[r] * D_ + dabs + lcol, acc[nt][1][r] * cwB[r]);
        }
    }
}

// ---------------------------------------------------------------- host
extern "C" void kernel_launch(void* const* d_in, const int* in_sizes, int n_in,
                              void* d_out, int out_size, void* d_ws, size_t ws_size,
                              hipStream_t stream) {
    const float* x      = (const float*)d_in[0];
    const float* rscale = (const float*)d_in[1];
    const float* rw     = (const float*)d_in[2];
    const float* gating = (const float*)d_in[3];
    const float* linear = (const float*)d_in[4];
    const float* pes    = (const float*)d_in[5];
    float* out = (float*)d_out;

    char* p = (char*)d_ws;
    auto carve = [&](size_t bytes) {
        char* q = p;
        p += (bytes + 255) & ~(size_t)255;
        return (void*)q;
    };
    u16*   xb      = (u16*)  carve((size_t)NTOK * D_ * sizeof(u16));
    u16*   wgb     = (u16*)  carve((size_t)E_ * 2 * HID_ * D_ * sizeof(u16));
    u16*   w2t     = (u16*)  carve((size_t)E_ * D_ * HID_ * sizeof(u16));
    u16*   act_ws  = (u16*)  carve((size_t)E_ * NTOK * HID_ * sizeof(u16));
    float* combine = (float*)carve((size_t)NTOK * E_ * sizeof(float));
    int*   counts  = (int*)  carve(E_ * sizeof(int));
    int*   idxlist = (int*)  carve((size_t)E_ * NTOK * sizeof(int));

    const size_t nOut4 = (size_t)NTOK * D_ / 4;
    k_zero<<<2048, 256, 0, stream>>>((uint4v*)out, nOut4, counts);
    k_cvt<<<8192, 256, 0, stream>>>(gating, wgb, (size_t)E_ * 2 * HID_ * D_);
    k_transpose_w2<<<dim3(E_ * (HID_ / 32), D_ / 32), 256, 0, stream>>>(linear, w2t);
    k_router<<<NTOK / 8, 256, 0, stream>>>(x, rscale, rw, pes, combine, counts, idxlist, xb);
    k_act<<<dim3(E_ * MT32, HID_ / 512), 256, 0, stream>>>(xb, wgb, counts, idxlist, act_ws);
    k_down<<<dim3(E_ * MT32, D_ / 512), 256, 0, stream>>>(act_ws, w2t, combine, counts, idxlist, out);
}
